// DecoderRNN_Attn_33758442947399
// MI455X (gfx1250) — matrix operand order repaired and checked by reading, not compiled, not run on hardware
//
#include <hip/hip_runtime.h>
#include <hip/hip_bf16.h>
#include <math.h>

#define BB 32
#define SS 64
#define TT 64
#define HH 1024
#define EE 512
#define VV 32000
#define GG 4096  /* 4*H */

#define NTILE 128   /* N columns per block */
#define KCH   32    /* K chunk per WMMA */

typedef __attribute__((ext_vector_type(16))) __bf16 v16bf;
typedef __attribute__((ext_vector_type(8)))  __bf16 v8bf;
typedef __attribute__((ext_vector_type(8)))  float  v8f;

typedef unsigned short WTile[NTILE][KCH];

__device__ __forceinline__ unsigned short f2bf(float f) {
    unsigned int u = __float_as_uint(f);
    u += 0x7FFFu + ((u >> 16) & 1u);          // round-to-nearest-even
    return (unsigned short)(u >> 16);
}
__device__ __forceinline__ float sigf(float x) { return 1.0f / (1.0f + expf(-x)); }

// ---- WMMA fragment loaders (layouts per CDNA5 ISA 7.12.2, wave32) ----
// A: 16x32 bf16 tile of X[M,K] row-major. lane<16: K in {0..7,16..23}; lane>=16: {8..15,24..31}
__device__ __forceinline__ v16bf load_frag_a(const unsigned short* A, int ld,
                                             int rowBase, int k0, int lane) {
    int row = rowBase + (lane & 15);
    int kh  = (lane >> 4) * 8;
    const v8bf* plo = (const v8bf*)(A + (size_t)row * ld + k0 + kh);
    const v8bf* phi = (const v8bf*)(A + (size_t)row * ld + k0 + 16 + kh);
    v8bf lo = *plo, hi = *phi;
    v16bf f;
#pragma unroll
    for (int i = 0; i < 8; ++i) { f[i] = lo[i]; f[i + 8] = hi[i]; }
    return f;
}
// B: 32x16 bf16 tile of W^T read from the LDS-staged tile: buf[col][k] for the
// wave's 16 columns. lanes<16: K {0..7,16..23}, lanes>=16: K {8..15,24..31}.
__device__ __forceinline__ v16bf load_frag_b_lds(const unsigned short (*buf)[KCH],
                                                 int colBase, int lane) {
    int col = colBase + (lane & 15);
    int kh  = (lane >> 4) * 8;
    const v8bf* p = (const v8bf*)&buf[col][kh];
    v8bf lo = p[0], hi = *(const v8bf*)&buf[col][kh + 16];
    v16bf f;
#pragma unroll
    for (int i = 0; i < 8; ++i) { f[i] = lo[i]; f[i + 8] = hi[i]; }
    return f;
}

// Async-DMA one 128x32 bf16 weight tile (8KB) global -> LDS. 512 16B segments,
// 2 per thread. Tracked by ASYNCcnt (2 instructions per wave).
__device__ __forceinline__ void stage_w(const unsigned short* W, int ldK, int nBlockBase,
                                        int k0, unsigned short (*buf)[KCH], int tid) {
#pragma unroll
    for (int j = 0; j < 2; ++j) {
        int s   = tid + j * 256;     // 0..511
        int row = s >> 2;            // 0..127 (W row = output col)
        int seg = s & 3;             // 16B segment within the 64B row chunk
        const void* g = (const void*)(W + (size_t)(nBlockBase + row) * ldK + k0 + seg * 8);
        unsigned int l = (unsigned int)(size_t)&buf[row][seg * 8];
        asm volatile("global_load_async_to_lds_b128 %0, %1, off"
                     :: "v"(l), "v"((unsigned long long)(size_t)g) : "memory");
    }
}

// One K-pass of C += A * W^T with LDS double-buffered async weight staging.
__device__ __forceinline__ void gemm_pass(const unsigned short* __restrict__ A,
                                          const unsigned short* __restrict__ W, int K,
                                          int m0, int nb, int wave, int lane, int tid,
                                          WTile* sW, v8f& acc0, v8f& acc1) {
    const int nChunks = K / KCH;
    stage_w(W, K, nb, 0, sW[0], tid);
    for (int c = 0; c < nChunks; ++c) {
        const int k0  = c * KCH;
        const int cur = c & 1;
        if (c + 1 < nChunks) {
            stage_w(W, K, nb, k0 + KCH, sW[cur ^ 1], tid);
            asm volatile("s_wait_asynccnt 2" ::: "memory");   // current tile done
        } else {
            asm volatile("s_wait_asynccnt 0" ::: "memory");
        }
        __syncthreads();
        v16bf a0 = load_frag_a(A, K, m0,      k0, lane);
        v16bf a1 = load_frag_a(A, K, m0 + 16, k0, lane);
        v16bf bf = load_frag_b_lds(sW[cur], wave * 16, lane);
        acc0 = __builtin_amdgcn_wmma_f32_16x16x32_bf16(false, a0, false, bf, (short)0, acc0, false, false);
        acc1 = __builtin_amdgcn_wmma_f32_16x16x32_bf16(false, a1, false, bf, (short)0, acc1, false, false);
        __syncthreads();   // reads of this buffer done before it is re-staged
    }
}

// ---- generic C = A1*W1^T (+ A2*W2^T) + b1 (+ b2), optional tanh, f32 + optional bf16 out
__global__ __launch_bounds__(256)
void gemm_xwt(const unsigned short* __restrict__ A1, const unsigned short* __restrict__ W1, int K1,
              const unsigned short* __restrict__ A2, const unsigned short* __restrict__ W2, int K2,
              const float* __restrict__ bias1, const float* __restrict__ bias2,
              float* __restrict__ C, unsigned short* __restrict__ Cbf,
              int N, int act) {
    __shared__ WTile sW[2];     // 2 x 8KB double buffer
    const int tid   = threadIdx.x;
    const int lane  = tid & 31;
    const int wave  = tid >> 5;
    const int nb    = blockIdx.x * NTILE;
    const int m0    = blockIdx.y * 32;

    v8f acc0 = {}; v8f acc1 = {};
    gemm_pass(A1, W1, K1, m0, nb, wave, lane, tid, sW, acc0, acc1);
    if (A2)
        gemm_pass(A2, W2, K2, m0, nb, wave, lane, tid, sW, acc0, acc1);

    const int col  = nb + wave * 16 + (lane & 15);
    const int roff = (lane >> 4) * 8;
    float bb = (bias1 ? bias1[col] : 0.f) + (bias2 ? bias2[col] : 0.f);
#pragma unroll
    for (int r = 0; r < 8; ++r) {
        float v0 = acc0[r] + bb, v1 = acc1[r] + bb;
        if (act == 1) { v0 = tanhf(v0); v1 = tanhf(v1); }
        int row0 = m0 + roff + r;
        int row1 = m0 + 16 + roff + r;
        C[(size_t)row0 * N + col] = v0;
        C[(size_t)row1 * N + col] = v1;
        if (Cbf) {
            Cbf[(size_t)row0 * N + col] = f2bf(v0);
            Cbf[(size_t)row1 * N + col] = f2bf(v1);
        }
    }
}

// ---- fp32 -> bf16 (strided sub-matrix) ----
__global__ void cvt_bf16(const float* __restrict__ src, unsigned short* __restrict__ dst,
                         long long rows, long long cols, long long src_ld, long long coloff) {
    long long n = rows * cols;
    for (long long i = (long long)blockIdx.x * blockDim.x + threadIdx.x; i < n;
         i += (long long)gridDim.x * blockDim.x) {
        long long r = i / cols, c = i % cols;
        dst[i] = f2bf(src[r * src_ld + coloff + c]);
    }
}

__global__ void init_state(const float* __restrict__ enc_h, const float* __restrict__ enc_c,
                           float* h0, float* c0, float* h1, float* c1,
                           unsigned short* h0_bf, unsigned short* h1_bf, float* attn_h) {
    int i = blockIdx.x * blockDim.x + threadIdx.x;
    if (i >= BB * HH) return;
    float a = enc_h[i], b = enc_h[BB * HH + i];
    h0[i] = a; h1[i] = b;
    h0_bf[i] = f2bf(a); h1_bf[i] = f2bf(b);
    c0[i] = enc_c[i]; c1[i] = enc_c[BB * HH + i];
    attn_h[i] = 0.f;
}

__global__ void build_x(const int* __restrict__ tgt, int t, const float* __restrict__ emb,
                        const float* __restrict__ attn_h, unsigned short* __restrict__ x_bf) {
    int i = blockIdx.x * blockDim.x + threadIdx.x;
    if (i >= BB * (EE + HH)) return;
    int b = i / (EE + HH), col = i % (EE + HH);
    float v = (col < EE) ? emb[(size_t)tgt[b * SS + t] * EE + col]
                         : attn_h[b * HH + (col - EE)];
    x_bf[i] = f2bf(v);
}

__global__ void lstm_cell(const float* __restrict__ gates, float* __restrict__ c,
                          float* __restrict__ h, unsigned short* __restrict__ h_bf) {
    int i = blockIdx.x * blockDim.x + threadIdx.x;
    if (i >= BB * HH) return;
    int b = i >> 10, n = i & (HH - 1);
    const float* g = gates + (size_t)b * GG;
    float gi = g[n], gf = g[HH + n], gg = g[2 * HH + n], go = g[3 * HH + n];
    float cc = sigf(gf) * c[i] + sigf(gi) * tanhf(gg);
    float hh = sigf(go) * tanhf(cc);
    c[i] = cc; h[i] = hh; h_bf[i] = f2bf(hh);
}

// scores[b,t] = sum_n va[n]*tanh(q[b,n]+encPre[t,b,n]); softmax; context = sum_t w*enc
__global__ __launch_bounds__(256)
void attention(const float* __restrict__ q, const float* __restrict__ encPre,
               const float* __restrict__ enc_out, const float* __restrict__ va,
               float* __restrict__ context) {
    int b = blockIdx.x, lane = threadIdx.x & 31, wave = threadIdx.x >> 5;
    __shared__ float s_sc[TT];
    __shared__ float s_m, s_d;
    for (int t = wave; t < TT; t += 8) {
        float sum = 0.f;
        const float* ep = encPre + ((size_t)t * BB + b) * HH;
        const float* qp = q + (size_t)b * HH;
        for (int n = lane; n < HH; n += 32) sum += va[n] * tanhf(qp[n] + ep[n]);
#pragma unroll
        for (int off = 16; off > 0; off >>= 1) sum += __shfl_xor(sum, off);
        if (lane == 0) s_sc[t] = sum;
    }
    __syncthreads();
    if (threadIdx.x == 0) {
        float m = -1e30f;
        for (int t = 0; t < TT; ++t) m = fmaxf(m, s_sc[t]);
        float d = 0.f;
        for (int t = 0; t < TT; ++t) d += expf(s_sc[t] - m);
        s_m = m; s_d = 1.0f / d;
    }
    __syncthreads();
    if (threadIdx.x < TT) s_sc[threadIdx.x] = expf(s_sc[threadIdx.x] - s_m) * s_d;
    __syncthreads();
    for (int n = threadIdx.x; n < HH; n += 256) {
        float acc = 0.f;
        for (int t = 0; t < TT; ++t) acc += s_sc[t] * enc_out[((size_t)t * BB + b) * HH + n];
        context[(size_t)b * HH + n] = acc;
    }
}

__global__ void build_cat2(const float* __restrict__ h1, const float* __restrict__ ctx,
                           unsigned short* __restrict__ cat_bf) {
    int i = blockIdx.x * blockDim.x + threadIdx.x;
    if (i >= BB * 2 * HH) return;
    int b = i >> 11, col = i & (2 * HH - 1);
    float v = (col < HH) ? h1[b * HH + col] : ctx[b * HH + col - HH];
    cat_bf[i] = f2bf(v);
}

__global__ __launch_bounds__(256)
void logsoftmax_argmax(const float* __restrict__ logits, float* __restrict__ out_logp,
                       int* __restrict__ out_word) {
    int b = blockIdx.x, tid = threadIdx.x;
    const float* row = logits + (size_t)b * VV;
    __shared__ float sm[256]; __shared__ int si[256]; __shared__ float sd[256];
    __shared__ float g_m, g_l;
    float m = -1e30f; int mi = 0;
    for (int v = tid; v < VV; v += 256) {
        float x = row[v];
        if (x > m) { m = x; mi = v; }
    }
    sm[tid] = m; si[tid] = mi; __syncthreads();
    for (int s = 128; s > 0; s >>= 1) {
        if (tid < s) {
            if (sm[tid + s] > sm[tid] || (sm[tid + s] == sm[tid] && si[tid + s] < si[tid])) {
                sm[tid] = sm[tid + s]; si[tid] = si[tid + s];
            }
        }
        __syncthreads();
    }
    if (tid == 0) { g_m = sm[0]; out_word[b] = si[0]; }
    __syncthreads();
    float d = 0.f;
    for (int v = tid; v < VV; v += 256) d += expf(row[v] - g_m);
    sd[tid] = d; __syncthreads();
    for (int s = 128; s > 0; s >>= 1) { if (tid < s) sd[tid] += sd[tid + s]; __syncthreads(); }
    if (tid == 0) g_l = logf(sd[0]);
    __syncthreads();
    float sub = g_m + g_l;
    for (int v = tid; v < VV; v += 256) out_logp[(size_t)b * VV + v] = row[v] - sub;
}

extern "C" void kernel_launch(void* const* d_in, const int* in_sizes, int n_in,
                              void* d_out, int out_size, void* d_ws, size_t ws_size,
                              hipStream_t stream) {
    const int*   tgt   = (const int*)d_in[0];
    const float* enc_h = (const float*)d_in[1];
    const float* enc_c = (const float*)d_in[2];
    const float* enc_o = (const float*)d_in[3];
    const float* emb   = (const float*)d_in[4];
    const float* W_ih0 = (const float*)d_in[5];
    const float* W_hh0 = (const float*)d_in[6];
    const float* b_ih0 = (const float*)d_in[7];
    const float* b_hh0 = (const float*)d_in[8];
    const float* W_ih1 = (const float*)d_in[9];
    const float* W_hh1 = (const float*)d_in[10];
    const float* b_ih1 = (const float*)d_in[11];
    const float* b_hh1 = (const float*)d_in[12];
    const float* Wa    = (const float*)d_in[13];
    const float* ba    = (const float*)d_in[14];
    const float* va    = (const float*)d_in[15];
    const float* Wc    = (const float*)d_in[16];
    const float* bc    = (const float*)d_in[17];
    const float* Wo    = (const float*)d_in[18];
    const float* bo    = (const float*)d_in[19];

    size_t off = 0;
    auto alloc = [&](size_t bytes) -> char* {
        char* p = (char*)d_ws + off;
        off += (bytes + 255) & ~(size_t)255;
        return p;
    };
    unsigned short* Wih0_bf = (unsigned short*)alloc((size_t)GG * 1536 * 2);
    unsigned short* Whh0_bf = (unsigned short*)alloc((size_t)GG * HH * 2);
    unsigned short* Wih1_bf = (unsigned short*)alloc((size_t)GG * HH * 2);
    unsigned short* Whh1_bf = (unsigned short*)alloc((size_t)GG * HH * 2);
    unsigned short* WaH_bf  = (unsigned short*)alloc((size_t)HH * HH * 2);
    unsigned short* WaE_bf  = (unsigned short*)alloc((size_t)HH * HH * 2);
    unsigned short* Wc_bf   = (unsigned short*)alloc((size_t)HH * 2 * HH * 2);
    unsigned short* Wo_bf   = (unsigned short*)alloc((size_t)VV * HH * 2);
    unsigned short* enc_bf  = (unsigned short*)alloc((size_t)TT * BB * HH * 2);
    float* encPre = (float*)alloc((size_t)TT * BB * HH * 4);
    float* gates0 = (float*)alloc((size_t)BB * GG * 4);
    float* gates1 = (float*)alloc((size_t)BB * GG * 4);
    float* h0 = (float*)alloc(BB * HH * 4);
    float* c0 = (float*)alloc(BB * HH * 4);
    float* h1 = (float*)alloc(BB * HH * 4);
    float* c1 = (float*)alloc(BB * HH * 4);
    float* attn_h = (float*)alloc(BB * HH * 4);
    float* qbuf   = (float*)alloc(BB * HH * 4);
    float* ctx    = (float*)alloc(BB * HH * 4);
    unsigned short* h0_bf = (unsigned short*)alloc(BB * HH * 2);
    unsigned short* h1_bf = (unsigned short*)alloc(BB * HH * 2);
    unsigned short* x_bf  = (unsigned short*)alloc(BB * (EE + HH) * 2);
    unsigned short* cat_bf = (unsigned short*)alloc(BB * 2 * HH * 2);
    unsigned short* ah_bf  = (unsigned short*)alloc(BB * HH * 2);
    float* logits = (float*)alloc((size_t)BB * VV * 4);

    float* out_logp = (float*)d_out;
    int*   out_word = (int*)(out_logp + (size_t)(SS - 1) * BB * VV);

    auto cvt = [&](const float* s, unsigned short* d, long long r, long long c,
                   long long ld, long long co) {
        long long n = r * c;
        cvt_bf16<<<dim3((unsigned)((n + 255) / 256)), 256, 0, stream>>>(s, d, r, c, ld, co);
    };
    // one-time (per launch) bf16 weight conversion: total ~110 MB -> L2-resident
    cvt(W_ih0, Wih0_bf, GG, 1536, 1536, 0);
    cvt(W_hh0, Whh0_bf, GG, HH, HH, 0);
    cvt(W_ih1, Wih1_bf, GG, HH, HH, 0);
    cvt(W_hh1, Whh1_bf, GG, HH, HH, 0);
    cvt(Wa, WaH_bf, HH, HH, 2 * HH, 0);
    cvt(Wa, WaE_bf, HH, HH, 2 * HH, HH);
    cvt(Wc, Wc_bf, HH, 2 * HH, 2 * HH, 0);
    cvt(Wo, Wo_bf, VV, HH, HH, 0);
    cvt(enc_o, enc_bf, (long long)TT * BB, HH, HH, 0);

    init_state<<<dim3(BB * HH / 256), 256, 0, stream>>>(enc_h, enc_c, h0, c0, h1, c1,
                                                        h0_bf, h1_bf, attn_h);
    // loop-invariant: encPre[t,b,:] = enc_out @ Wa[:,H:]^T + ba   (M=2048,N=1024,K=1024)
    gemm_xwt<<<dim3(HH / NTILE, (TT * BB) / 32), 256, 0, stream>>>(
        enc_bf, WaE_bf, HH, nullptr, nullptr, 0, ba, nullptr, encPre, nullptr, HH, 0);

    for (int t = 0; t < SS - 1; ++t) {
        build_x<<<dim3((BB * (EE + HH) + 255) / 256), 256, 0, stream>>>(tgt, t, emb, attn_h, x_bf);
        // LSTM layer 0 gates: x@Wih0^T + h0@Whh0^T + b
        gemm_xwt<<<dim3(GG / NTILE, 1), 256, 0, stream>>>(
            x_bf, Wih0_bf, EE + HH, h0_bf, Whh0_bf, HH, b_ih0, b_hh0, gates0, nullptr, GG, 0);
        lstm_cell<<<dim3(BB * HH / 256), 256, 0, stream>>>(gates0, c0, h0, h0_bf);
        // LSTM layer 1 gates: h0@Wih1^T + h1_prev@Whh1^T + b
        gemm_xwt<<<dim3(GG / NTILE, 1), 256, 0, stream>>>(
            h0_bf, Wih1_bf, HH, h1_bf, Whh1_bf, HH, b_ih1, b_hh1, gates1, nullptr, GG, 0);
        lstm_cell<<<dim3(BB * HH / 256), 256, 0, stream>>>(gates1, c1, h1, h1_bf);
        // attention query: q = h1 @ Wa[:, :H]^T
        gemm_xwt<<<dim3(HH / NTILE, 1), 256, 0, stream>>>(
            h1_bf, WaH_bf, HH, nullptr, nullptr, 0, nullptr, nullptr, qbuf, nullptr, HH, 0);
        attention<<<dim3(BB), 256, 0, stream>>>(qbuf, encPre, enc_o, va, ctx);
        build_cat2<<<dim3(BB * 2 * HH / 256), 256, 0, stream>>>(h1, ctx, cat_bf);
        // attn_h = tanh([h1,ctx] @ Wc^T + bc)  (also emit bf16 copy for the Wo GEMM)
        gemm_xwt<<<dim3(HH / NTILE, 1), 256, 0, stream>>>(
            cat_bf, Wc_bf, 2 * HH, nullptr, nullptr, 0, bc, nullptr, attn_h, ah_bf, HH, 1);
        // logits = attn_h @ Wo^T + bo  (N=32000 -> 250 blocks, Wo streamed from L2 via async DMA)
        gemm_xwt<<<dim3(VV / NTILE, 1), 256, 0, stream>>>(
            ah_bf, Wo_bf, HH, nullptr, nullptr, 0, bo, nullptr, logits, nullptr, VV, 0);
        logsoftmax_argmax<<<dim3(BB), 256, 0, stream>>>(
            logits, out_logp + (size_t)t * BB * VV, out_word + (size_t)t * BB);
    }
    (void)in_sizes; (void)n_in; (void)out_size; (void)ws_size;
}
